// SceneCollisionEncoder_38500086841674
// MI455X (gfx1250) — matrix operand order repaired and checked by reading, not compiled, not run on hardware
//
#include <hip/hip_runtime.h>

// ---------------------------------------------------------------------------
// MI455X (gfx1250, wave32) implementation.
// Compute-bound pipeline (~0.72 TFLOP vs ~10us of HBM traffic at 23.3 TB/s),
// so all GEMM-able work is routed through v_wmma_f32_16x16x32_f16 (f32 acc).
// Activations/weights are converted to f16; accumulation stays in f32.
// ---------------------------------------------------------------------------

typedef __attribute__((ext_vector_type(16))) _Float16 v16h;
typedef __attribute__((ext_vector_type(8)))  float    v8f;
typedef __attribute__((ext_vector_type(4)))  float    f32x4;
typedef __attribute__((ext_vector_type(4)))  int      i32x4;

union frag16 { v16h h; f32x4 f4[2]; };

#define WMMA_F16(a, b, c) \
  __builtin_amdgcn_wmma_f32_16x16x32_f16(false, (a), false, (b), (short)0, (c), false, false)

// Async global->LDS (CDNA5 ASYNCcnt path); guarded so either toolchain compiles.
#if defined(__HIP_DEVICE_COMPILE__) && \
    __has_builtin(__builtin_amdgcn_global_load_async_to_lds_b128) && \
    __has_builtin(__builtin_amdgcn_s_wait_asynccnt)
#define HAVE_ASYNC_LDS 1
#else
#define HAVE_ASYNC_LDS 0
#endif

#define AS_GLOBAL __attribute__((address_space(1)))
#define AS_LOCAL  __attribute__((address_space(3)))

// Problem constants
#define BATCH 2
#define NPTS  100000
#define NVOX  32
#define VVOX  (NVOX * NVOX * NVOX)   // 32768

// Workspace layout (bytes)
#define OFF_GRID   ((size_t)0)            // u32/f32 (B,256,V) atomic-max grid : 67108864
#define OFF_GRIDH  ((size_t)67108864)     // f16 (B,V,256) NDHWC              : 33554432
#define OFF_L1H    ((size_t)100663296)    // f16 (B,4096,512)                 : 8388608
#define OFF_L2H    ((size_t)109051904)    // f16 (B,512,1024)                 : 2097152
#define OFF_W2H    ((size_t)111149056)    // f16 (256,128)                    : 65536
#define OFF_WC1H   ((size_t)111214592)    // f16 (512,27,256)                 : 7077888
#define OFF_WC2H   ((size_t)118292480)    // f16 (1024,27,512)                : 28311552
#define OFF_WTH    ((size_t)146604032)    // f16 (8,512,1024)                 : 8388608

// ---------------------------------------------------------------------------
// Weight conversion / relayout kernels (tiny; run once per launch)
// ---------------------------------------------------------------------------
__global__ void cvt_copy_f16(const float* __restrict__ in, _Float16* __restrict__ out, int n) {
  int i = blockIdx.x * blockDim.x + threadIdx.x;
  if (i < n) out[i] = (_Float16)in[i];
}

// (OC,IC,3,3,3) -> (OC,27,IC), tap t = kz*9+ky*3+kx
__global__ void cvt_wc_f16(const float* __restrict__ in, _Float16* __restrict__ out,
                           int IC, int n) {
  int d = blockIdx.x * blockDim.x + threadIdx.x;
  if (d >= n) return;
  int ic = d % IC;
  int t  = (d / IC) % 27;
  int oc = d / (IC * 27);
  out[d] = (_Float16)in[((size_t)oc * IC + ic) * 27 + t];
}

// Wt (1024,512,2,2,2) -> (tap, oc=512, ic=1024); tap = (tz*2+ty)*2+tx
__global__ void cvt_wt_f16(const float* __restrict__ in, _Float16* __restrict__ out) {
  int d = blockIdx.x * blockDim.x + threadIdx.x;   // 8*512*1024
  int ic = d & 1023, oc = (d >> 10) & 511, t = d >> 19;
  out[d] = (_Float16)in[(((size_t)ic * 512 + oc) << 3) + t];
}

// ---------------------------------------------------------------------------
// Kernel 1: per-point MLP (3->128 VALU, 128->256 WMMA) + voxel atomic max.
// 8 waves/block, wave handles 16 points.  Grid zero-init + u32 atomicMax on
// non-negative float bits == segment_max semantics of the reference.
// ---------------------------------------------------------------------------
__global__ __launch_bounds__(256) void mlp_scatter_kernel(
    const float* __restrict__ pc, const float* __restrict__ W1,
    const float* __restrict__ b1, const _Float16* __restrict__ W2h,
    const float* __restrict__ b2, unsigned int* __restrict__ grid) {
  __shared__ float w1s[384];
  __shared__ float b1s[128];
  __shared__ float b2s[256];
  __shared__ __align__(16) _Float16 Hs[8][16][128];   // per-wave h, K-contiguous

  int tid = threadIdx.x;
  for (int i = tid; i < 384; i += 256) w1s[i] = W1[i];
  if (tid < 128) b1s[tid] = b1[tid];
  b2s[tid] = b2[tid];
  __syncthreads();

  int w = tid >> 5, lane = tid & 31, r = lane & 15, hi = lane >> 4;
  int b = blockIdx.y;
  int p = blockIdx.x * 128 + w * 16 + r;
  bool valid = p < NPTS;

  float x = 0.f, y = 0.f, z = 0.f;
  if (valid) {
    const float* q = pc + ((size_t)b * NPTS + p) * 3;
    x = q[0]; y = q[1]; z = q[2];
  }
  int ix = min(31, max(0, (int)floorf((x + 1.0f) * 16.0f)));
  int iy = min(31, max(0, (int)floorf((y + 1.0f) * 16.0f)));
  int iz = min(31, max(0, (int)floorf((z + 1.0f) * 16.0f)));
  int flat = (ix * 32 + iy) * 32 + iz;
  float xc0 = x - (ix * 0.0625f + 0.03125f - 1.0f);
  float xc1 = y - (iy * 0.0625f + 0.03125f - 1.0f);
  float xc2 = z - (iz * 0.0625f + 0.03125f - 1.0f);

  // layer 1: each half-wave computes 64 of the 128 h values for its point
  for (int row = hi * 64; row < hi * 64 + 64; ++row) {
    float hv = w1s[row * 3] * xc0 + w1s[row * 3 + 1] * xc1 + w1s[row * 3 + 2] * xc2 + b1s[row];
    Hs[w][r][row] = (_Float16)fmaxf(hv, 0.0f);
  }
  __syncthreads();

  unsigned int* gbase = grid + (size_t)b * 256 * VVOX;
  for (int ot = 0; ot < 16; ++ot) {          // 256 outputs / 16
    v8f c;
#pragma unroll
    for (int j = 0; j < 8; ++j) c[j] = b2s[ot * 16 + j + hi * 8];
#pragma unroll
    for (int kb = 0; kb < 4; ++kb) {         // K = 128 / 32
      int ks = kb * 32;
      frag16 a, bf;
      const _Float16* ap = W2h + (size_t)(ot * 16 + r) * 128 + ks + hi * 8;
      a.f4[0] = *(const f32x4*)(ap);
      a.f4[1] = *(const f32x4*)(ap + 16);
      const _Float16* bp = &Hs[w][r][ks + hi * 8];
      bf.f4[0] = *(const f32x4*)(bp);
      bf.f4[1] = *(const f32x4*)(bp + 16);
      c = WMMA_F16(a.h, bf.h, c);
    }
    if (valid) {
#pragma unroll
      for (int j = 0; j < 8; ++j) {
        float v = fmaxf(c[j], 0.0f);
        int oc = ot * 16 + j + hi * 8;
        atomicMax(gbase + (size_t)oc * VVOX + flat, __float_as_uint(v));
      }
    }
  }
}

// ---------------------------------------------------------------------------
// Kernel 2: grid (B,256,V) f32 -> (B,V,256) f16 NDHWC for implicit GEMM
// ---------------------------------------------------------------------------
__global__ __launch_bounds__(256) void grid_to_ndhwc(const float* __restrict__ g,
                                                     _Float16* __restrict__ gh) {
  int t = blockIdx.x * blockDim.x + threadIdx.x;   // B*V threads
  int b = t >> 15, s = t & (VVOX - 1);
  const float* src = g + (size_t)b * 256 * VVOX + s;
  _Float16* dst = gh + (size_t)t * 256;
  for (int c8 = 0; c8 < 32; ++c8) {
    union { f32x4 v; _Float16 e[8]; } u;
#pragma unroll
    for (int k = 0; k < 8; ++k) u.e[k] = (_Float16)src[(size_t)(c8 * 8 + k) * VVOX];
    *(f32x4*)(dst + c8 * 8) = u.v;
  }
}

// ---------------------------------------------------------------------------
// Kernel 3/4 (template): conv3d(3x3x3, pad 1) + relu + maxpool(2) implicit GEMM
//   K order = tap-major (K = tap*IC + ic) -> activations contiguous in NDHWC.
//   Block: 8 waves x 16 oc = 128 oc, spatial tile (2,2,4) -> pooled (1,1,2).
//   Halo tile (4x4x6 x <=256ch) staged in LDS via async global->LDS.
// ---------------------------------------------------------------------------
template <int NVD, int IC, int OC, bool OUT32>
__global__ __launch_bounds__(256) void conv3_relu_pool_kernel(
    const _Float16* __restrict__ inh,   // (B, NVD^3, IC)
    const _Float16* __restrict__ Wh,    // (OC, 27, IC)
    const float* __restrict__ bias,     // (OC)
    float* __restrict__ out32,          // d_out (B,1024,P^3) channels [0,OC)
    _Float16* __restrict__ outh) {      // (B, P^3, OC)
  constexpr int ICP = (IC > 256) ? 256 : IC;
  constexpr int PASSES = IC / ICP;
  constexpr int TILESX = NVD / 4, TILESY = NVD / 2;
  constexpr int P = NVD / 2;

  __shared__ __align__(16) _Float16 tile[4 * 4 * 6 * ICP];  // 48KB max
  __shared__ float poolbuf[8][16][16];

  int tid = threadIdx.x;
  int w = tid >> 5, lane = tid & 31, r = lane & 15, hi = lane >> 4;
  int b = blockIdx.z;
  int s = blockIdx.x;
  int tx = s % TILESX, ty = (s / TILESX) % TILESY, tz = s / (TILESX * TILESY);
  int x0 = tx * 4, y0 = ty * 2, z0 = tz * 2;
  int ocBase = blockIdx.y * 128 + w * 16;
  const _Float16* inB = inh + (size_t)b * NVD * NVD * NVD * IC;

  int dxo = r & 3, dyo = (r >> 2) & 1, dzo = r >> 3;   // this lane's output pos

  v8f c;
#pragma unroll
  for (int j = 0; j < 8; ++j) c[j] = bias[ocBase + j + hi * 8];

  for (int pass = 0; pass < PASSES; ++pass) {
    int icBase = pass * ICP;
    __syncthreads();   // previous-pass tile reads done
    // cooperative halo-tile load: 96 positions x ICP channels, 16B chunks
    for (int i = tid; i < 96 * (ICP / 8); i += 256) {
      int pos = i / (ICP / 8);
      int c8  = (i % (ICP / 8)) * 8;
      int dx = pos % 6, dy = (pos / 6) % 4, dz = pos / 24;
      int gx = x0 - 1 + dx, gy = y0 - 1 + dy, gz = z0 - 1 + dz;
      _Float16* dst = &tile[pos * ICP + c8];
      if (gx >= 0 && gx < NVD && gy >= 0 && gy < NVD && gz >= 0 && gz < NVD) {
        const _Float16* src = inB + (size_t)((gz * NVD + gy) * NVD + gx) * IC + icBase + c8;
#if HAVE_ASYNC_LDS
        __builtin_amdgcn_global_load_async_to_lds_b128(
            (AS_GLOBAL i32x4*)(AS_GLOBAL void*)(void*)src,
            (AS_LOCAL  i32x4*)(AS_LOCAL  void*)dst, 0, 0);
#else
        *(f32x4*)dst = *(const f32x4*)src;
#endif
      } else {
        f32x4 zz = {0.f, 0.f, 0.f, 0.f};
        *(f32x4*)dst = zz;
      }
    }
#if HAVE_ASYNC_LDS
    __builtin_amdgcn_s_wait_asynccnt(0);
#endif
    __syncthreads();

    for (int t = 0; t < 27; ++t) {
      int kx = t % 3, ky = (t / 3) % 3, kz = t / 9;
      int ldsBase = (((dzo + kz) * 4 + (dyo + ky)) * 6 + (dxo + kx)) * ICP;
      const _Float16* aBase = Wh + ((size_t)(ocBase + r) * 27 + t) * IC + icBase;
      __builtin_prefetch(aBase + IC, 0, 1);   // next tap's weight row
#pragma unroll
      for (int cb = 0; cb < ICP / 32; ++cb) {
        int cs = cb * 32;
        frag16 a, bf;
        const _Float16* ap = aBase + cs + hi * 8;
        a.f4[0] = *(const f32x4*)(ap);
        a.f4[1] = *(const f32x4*)(ap + 16);
        const _Float16* bp = &tile[ldsBase + cs + hi * 8];
        bf.f4[0] = *(const f32x4*)(bp);
        bf.f4[1] = *(const f32x4*)(bp + 16);
        c = WMMA_F16(a.h, bf.h, c);
      }
    }
  }

  // relu + 2x2x2 maxpool (cross-lane via LDS)
#pragma unroll
  for (int j = 0; j < 8; ++j) poolbuf[w][j + hi * 8][r] = fmaxf(c[j], 0.0f);
  __syncthreads();

  int oco = lane >> 1, pxo = lane & 1;
  float m = 0.0f;   // relu'd values are >= 0
#pragma unroll
  for (int dz2 = 0; dz2 < 2; ++dz2)
#pragma unroll
    for (int dy2 = 0; dy2 < 2; ++dy2)
#pragma unroll
      for (int dx2 = 0; dx2 < 2; ++dx2) {
        int n = (dz2 * 2 + dy2) * 4 + (pxo * 2 + dx2);
        m = fmaxf(m, poolbuf[w][oco][n]);
      }
  int oc = ocBase + oco;
  int px = tx * 2 + pxo;
  int sp = (tz * P + ty) * P + px;
  if (OUT32) out32[((size_t)b * 1024 + oc) * (P * P * P) + sp] = m;
  outh[((size_t)b * (P * P * P) + sp) * OC + oc] = (_Float16)m;
}

// ---------------------------------------------------------------------------
// Kernel 5: convT(1024->512, 2x2x2, stride 2).  Non-overlapping: each output
// voxel = one tap GEMM.  out[.., 2i+t] = sum_ic Wt[ic][oc][t] * l2[ic][i].
// ---------------------------------------------------------------------------
__global__ __launch_bounds__(256) void convt_kernel(
    const _Float16* __restrict__ l2h,   // (B, 512, 1024)
    const _Float16* __restrict__ Wth,   // (8, 512, 1024)
    const float* __restrict__ bt,       // (512)
    float* __restrict__ out) {          // d_out, channels [512,1024)
  int tid = threadIdx.x;
  int w = tid >> 5, lane = tid & 31, r = lane & 15, hi = lane >> 4;
  int bx = blockIdx.x;
  int tile = bx & 31, tap = bx >> 5;
  int b = blockIdx.z;
  int ocBase = blockIdx.y * 128 + w * 16;

  int p = tile * 16 + r;                       // input (pooled) position
  int iz = p >> 6, iy = (p >> 3) & 7, ix = p & 7;
  int tzp = tap >> 2, typ = (tap >> 1) & 1, txp = tap & 1;

  v8f c;
#pragma unroll
  for (int j = 0; j < 8; ++j) c[j] = bt[ocBase + j + hi * 8];

  const _Float16* bBase = l2h + (size_t)(b * 512 + p) * 1024;
  const _Float16* aBase = Wth + (size_t)(tap * 512 + ocBase + r) * 1024;
  for (int kb = 0; kb < 32; ++kb) {            // K = 1024 / 32
    int ks = kb * 32;
    frag16 a, bf;
    __builtin_prefetch(aBase + ks + 128, 0, 1);
    a.f4[0] = *(const f32x4*)(aBase + ks + hi * 8);
    a.f4[1] = *(const f32x4*)(aBase + ks + hi * 8 + 16);
    bf.f4[0] = *(const f32x4*)(bBase + ks + hi * 8);
    bf.f4[1] = *(const f32x4*)(bBase + ks + hi * 8 + 16);
    c = WMMA_F16(a.h, bf.h, c);
  }

  int oz = 2 * iz + tzp, oy = 2 * iy + typ, ox = 2 * ix + txp;
  int sp = (oz * 16 + oy) * 16 + ox;
  float* ob = out + ((size_t)b * 1024 + 512 + ocBase) * 4096;
#pragma unroll
  for (int j = 0; j < 8; ++j) ob[(size_t)(j + hi * 8) * 4096 + sp] = c[j];
}

// ---------------------------------------------------------------------------
extern "C" void kernel_launch(void* const* d_in, const int* in_sizes, int n_in,
                              void* d_out, int out_size, void* d_ws, size_t ws_size,
                              hipStream_t stream) {
  const float* pc  = (const float*)d_in[0];
  const float* W1  = (const float*)d_in[1];
  const float* b1  = (const float*)d_in[2];
  const float* W2  = (const float*)d_in[3];
  const float* b2  = (const float*)d_in[4];
  const float* Wc1 = (const float*)d_in[5];
  const float* bc1 = (const float*)d_in[6];
  const float* Wc2 = (const float*)d_in[7];
  const float* bc2 = (const float*)d_in[8];
  const float* Wt  = (const float*)d_in[9];
  const float* bt  = (const float*)d_in[10];
  float* out = (float*)d_out;
  char* ws = (char*)d_ws;

  unsigned int* grid = (unsigned int*)(ws + OFF_GRID);
  _Float16* gridh = (_Float16*)(ws + OFF_GRIDH);
  _Float16* l1h   = (_Float16*)(ws + OFF_L1H);
  _Float16* l2h   = (_Float16*)(ws + OFF_L2H);
  _Float16* W2h   = (_Float16*)(ws + OFF_W2H);
  _Float16* Wc1h  = (_Float16*)(ws + OFF_WC1H);
  _Float16* Wc2h  = (_Float16*)(ws + OFF_WC2H);
  _Float16* Wth   = (_Float16*)(ws + OFF_WTH);

  // zero the atomic-max grid (empty voxel == 0.0f, matches reference)
  (void)hipMemsetAsync(grid, 0, (size_t)BATCH * 256 * VVOX * 4, stream);

  // weight conversion / relayout to f16
  cvt_copy_f16<<<(32768 + 255) / 256, 256, 0, stream>>>(W2, W2h, 32768);
  {
    int n = 512 * 27 * 256;
    cvt_wc_f16<<<(n + 255) / 256, 256, 0, stream>>>(Wc1, Wc1h, 256, n);
  }
  {
    int n = 1024 * 27 * 512;
    cvt_wc_f16<<<(n + 255) / 256, 256, 0, stream>>>(Wc2, Wc2h, 512, n);
  }
  cvt_wt_f16<<<(8 * 512 * 1024) / 256, 256, 0, stream>>>(Wt, Wth);

  // point MLP + voxel scatter-max
  mlp_scatter_kernel<<<dim3((NPTS + 127) / 128, BATCH), 256, 0, stream>>>(
      pc, W1, b1, W2h, b2, grid);

  // transpose grid to NDHWC f16
  grid_to_ndhwc<<<(BATCH * VVOX) / 256, 256, 0, stream>>>((const float*)grid, gridh);

  // conv1 + relu + pool -> l1 (f32 into d_out[:,0:512], f16 into ws)
  conv3_relu_pool_kernel<32, 256, 512, true>
      <<<dim3(2048, 4, BATCH), 256, 0, stream>>>(gridh, Wc1h, bc1, out, l1h);

  // conv2 + relu + pool -> l2 (f16 only)
  conv3_relu_pool_kernel<16, 512, 1024, false>
      <<<dim3(256, 8, BATCH), 256, 0, stream>>>(l1h, Wc2h, bc2, nullptr, l2h);

  // convT -> d_out[:,512:1024]
  convt_kernel<<<dim3(256, 4, BATCH), 256, 0, stream>>>(l2h, Wth, bt, out);
}
